// EfficientRelationConstructor_53360673686164
// MI455X (gfx1250) — compile-verified, matrix operand order checked by reading, not run on hardware
//
#include <hip/hip_runtime.h>

typedef __attribute__((ext_vector_type(16))) _Float16 v16h;
typedef __attribute__((ext_vector_type(8)))  _Float16 v8h;
typedef __attribute__((ext_vector_type(8)))  float    v8f;

#define BATCH 16
#define CH    256
#define NTOK  1600
#define NT    100                      // 16-wide tiles per row/col
#define SROW  1612                     // padded LDS row stride (floats); 1612%64=12
#define INV_T 14.285714285714286f      // 1 / 0.07

// ---------------------------------------------------------------------------
// Kernel 1: [B, C, N] f32 -> L2-normalized [B, N, C] f16 rows in workspace.
// Reads coalesced across threads (consecutive n); writes vectorized 16B.
// ---------------------------------------------------------------------------
__global__ __launch_bounds__(256) void norm_pack_kernel(
    const float* __restrict__ in, _Float16* __restrict__ ws) {
  int g = blockIdx.x * 256 + threadIdx.x;   // 0 .. B*N-1 (exactly 100 blocks)
  int b = g / NTOK;
  int n = g % NTOK;
  const float* src = in + (size_t)b * CH * NTOK + n;
  float acc = 0.f;
  for (int c = 0; c < CH; ++c) {
    float v = src[(size_t)c * NTOK];
    acc += v * v;
  }
  float inv = 1.0f / fmaxf(sqrtf(acc), 1e-12f);
  _Float16* dst = ws + ((size_t)b * NTOK + n) * CH;
  for (int c0 = 0; c0 < CH; c0 += 8) {
    v8h pack;
#pragma unroll
    for (int i = 0; i < 8; ++i)
      pack[i] = (_Float16)(src[(size_t)(c0 + i) * NTOK] * inv);
    *(v8h*)(dst + c0) = pack;
  }
}

// Load 8 K-chunks (K=256) of a 16x256 f16 tile into WMMA A/B fragments.
// ISA 16-bit 16x32 layout: lane L -> row (L&15); lanes 0-15 cover K-octets
// {0..7,16..23}, lanes 16-31 cover {8..15,24..31}: two 16-byte loads/chunk.
__device__ inline void load_frags(const _Float16* __restrict__ tile,
                                  int lane, v16h* frag) {
  const _Float16* rp = tile + (lane & 15) * CH + ((lane >> 4) * 8);
#pragma unroll
  for (int c = 0; c < 8; ++c) {
    v8h lo = *(const v8h*)(rp + 32 * c);
    v8h hi = *(const v8h*)(rp + 32 * c + 16);
    v16h f;
#pragma unroll
    for (int i = 0; i < 8; ++i) { f[i] = lo[i]; f[i + 8] = hi[i]; }
    frag[c] = f;
  }
}

// 8 WMMAs (K=256) with two independent accumulator chains, then write the
// 16x16 f32 block into the padded LDS stripe.
__device__ inline void mma_store(const v16h* afrag, const v16h* bfrag,
                                 float* __restrict__ simbuf, int ct, int lane) {
  v8f acc0 = {}, acc1 = {};
#pragma unroll
  for (int k = 0; k < 8; k += 2) {
    acc0 = __builtin_amdgcn_wmma_f32_16x16x32_f16(
        false, afrag[k],     false, bfrag[k],     (short)0, acc0, false, false);
    acc1 = __builtin_amdgcn_wmma_f32_16x16x32_f16(
        false, afrag[k + 1], false, bfrag[k + 1], (short)0, acc1, false, false);
  }
  // C/D layout: VGPR r -> M = r + 8*(lane/16), N = lane&15.
  int m0 = (lane >> 4) * 8;
  int n  = lane & 15;
#pragma unroll
  for (int r = 0; r < 8; ++r)
    simbuf[(m0 + r) * SROW + ct * 16 + n] = acc0[r] + acc1[r];
}

// ---------------------------------------------------------------------------
// Kernel 2: fused WMMA Gram-matrix + row softmax.
// One workgroup = one (batch, 16-row stripe); 8 waves split the 100 column
// tiles with double-buffered B fragments; the 16x1612 f32 stripe lives in
// LDS (103.2 KB < 320 KB/WGP).
// ---------------------------------------------------------------------------
__global__ __launch_bounds__(256) void relation_kernel(
    const _Float16* __restrict__ ws, float* __restrict__ out) {
  __shared__ float simbuf[16 * SROW];   // 103168 bytes

  int b    = blockIdx.x / NT;
  int rt   = blockIdx.x % NT;
  int lane = threadIdx.x & 31;
  // Scalarize the wave id so loop control is SALU and EXEC stays all-ones.
  int wave = __builtin_amdgcn_readfirstlane((int)(threadIdx.x >> 5));

  const _Float16* base = ws + (size_t)b * NTOK * CH;

  // A fragments for this 16-row stripe stay in registers for all 100 tiles.
  v16h afrag[8];
  load_frags(base + (size_t)rt * 16 * CH, lane, afrag);

  // Double-buffered column loop, 2x unrolled (no dynamic frag indexing).
  v16h bbuf0[8], bbuf1[8];
  int ct = wave;
  load_frags(base + (size_t)ct * 16 * CH, lane, bbuf0);
  while (true) {
    int ct1 = ct + 8;
    bool more1 = ct1 < NT;
    if (more1) load_frags(base + (size_t)ct1 * 16 * CH, lane, bbuf1);
    mma_store(afrag, bbuf0, simbuf, ct, lane);
    if (!more1) break;

    int ct2 = ct1 + 8;
    bool more2 = ct2 < NT;
    if (more2) load_frags(base + (size_t)ct2 * 16 * CH, lane, bbuf0);
    mma_store(afrag, bbuf1, simbuf, ct1, lane);
    if (!more2) break;
    ct = ct2;
  }
  __syncthreads();

  // Row softmax: 16 threads per row (half-wave), strided over 1600 columns.
  int row = threadIdx.x >> 4;           // 0..15
  int j   = threadIdx.x & 15;
  float* srow = simbuf + row * SROW;

  float m = -3.4e38f;
  for (int i = j; i < NTOK; i += 16) m = fmaxf(m, srow[i]);
#pragma unroll
  for (int off = 8; off; off >>= 1) m = fmaxf(m, __shfl_xor(m, off, 16));

  float s = 0.f;
  for (int i = j; i < NTOK; i += 16) {
    float e = __expf((srow[i] - m) * INV_T);
    srow[i] = e;
    s += e;
  }
#pragma unroll
  for (int off = 8; off; off >>= 1) s += __shfl_xor(s, off, 16);
  float invs = 1.0f / s;

  // Coalesced stream-out: threads j=0..15 cover 16 consecutive columns.
  float* orow = out + ((size_t)b * NTOK + rt * 16 + row) * NTOK;
  for (int i = j; i < NTOK; i += 16) orow[i] = srow[i] * invs;
}

extern "C" void kernel_launch(void* const* d_in, const int* in_sizes, int n_in,
                              void* d_out, int out_size, void* d_ws, size_t ws_size,
                              hipStream_t stream) {
  const float* in = (const float*)d_in[0];
  float* out      = (float*)d_out;
  _Float16* ws    = (_Float16*)d_ws;   // needs 16*1600*256*2 = 13.1 MB

  norm_pack_kernel<<<(BATCH * NTOK) / 256, 256, 0, stream>>>(in, ws);
  relation_kernel<<<BATCH * NT, 256, 0, stream>>>(ws, out);
}
